// TTSModel_62079457296568
// MI455X (gfx1250) — compile-verified, matrix operand order checked by reading, not run on hardware
//
#include <hip/hip_runtime.h>
#include <math.h>

#define DEV static __device__ __forceinline__

typedef __attribute__((ext_vector_type(16))) __bf16 v16bf;
typedef __attribute__((ext_vector_type(8)))  float  v8f;
typedef __attribute__((ext_vector_type(4)))  unsigned int u32x4;
typedef __bf16 bf16;

union AFrag { v16bf v; bf16 e[16]; u32x4 q[2]; };
union CFrag { v8f v; float f[8]; };

// CDNA5 16-bit fragment: lanes 0-15 hold K {0..7,16..23}, lanes 16-31 hold
// K {8..15,24..31}.  With a contiguous bf16 row this is two 16B loads.
DEV v16bf load_frag16(const bf16* p, int lane) {
  p += (lane & 16) ? 8 : 0;
  AFrag u;
  u.q[0] = *(const u32x4*)(p);
  u.q[1] = *(const u32x4*)(p + 16);
  return u.v;
}

DEV v8f wmma_bf16(v16bf a, v16bf b, v8f c) {
  return __builtin_amdgcn_wmma_f32_16x16x32_bf16(false, a, false, b, (short)0, c,
                                                 false, false);
}

// A row and B row both contiguous bf16, K multiple of 32.
DEV CFrag gemm_fast(const bf16* arow, const bf16* brow, int K, int lane) {
  CFrag c;
#pragma unroll
  for (int i = 0; i < 8; ++i) c.f[i] = 0.0f;
  for (int k0 = 0; k0 < K; k0 += 32)
    c.v = wmma_bf16(load_frag16(arow + k0, lane), load_frag16(brow + k0, lane), c.v);
  return c;
}

DEV float sigm(float x) { return 1.0f / (1.0f + expf(-x)); }

// ---------------- conversion / packing prep kernels ----------------

__global__ void k_cvt_pad(const float* __restrict__ src, bf16* __restrict__ dst,
                          int cols, int colsPad) {
  int r = blockIdx.y;
  int k = blockIdx.x * 256 + threadIdx.x;
  if (k < colsPad)
    dst[(size_t)r * colsPad + k] =
        (k < cols) ? (bf16)src[(size_t)r * cols + k] : (bf16)0.0f;
}

// encoder conv weight [512][512][5] -> bf16 [512][k=kt*512+cin]
__global__ void k_cvt_convw(const float* __restrict__ W, bf16* __restrict__ dst) {
  int m = blockIdx.y;
  int k = blockIdx.x * 256 + threadIdx.x;
  if (k < 2560) {
    int kt = k >> 9, cin = k & 511;
    dst[m * 2560 + k] = (bf16)W[m * 2560 + cin * 5 + kt];
  }
}

__global__ void k_embed(const int* __restrict__ tok, const float* __restrict__ emb,
                        float* __restrict__ x0) {
  int idx = blockIdx.x * blockDim.x + threadIdx.x;
  if (idx >= 8 * 512 * 128) return;
  int b = idx >> 16, d = (idx >> 7) & 511, t = idx & 127;
  x0[idx] = emb[tok[b * 128 + t] * 512 + d];  // [b][d][t]
}

// act f32 [B][512][128] -> im2col bf16 [n=(b,t)][k=kt*512+cin], zero padded
__global__ void k_im2col_enc(const float* __restrict__ act, bf16* __restrict__ colb) {
  int n = blockIdx.y;
  int b = n >> 7, t = n & 127;
  int k = blockIdx.x * 256 + threadIdx.x;
  if (k >= 2560) return;
  int kt = k >> 9, cin = k & 511, tt = t + kt - 2;
  colb[n * 2560 + k] =
      (tt >= 0 && tt < 128) ? (bf16)act[(b * 512 + cin) * 128 + tt] : (bf16)0.0f;
}

// act3 f32 [B][512][128] -> xT bf16 [t][16][512] (rows 8-15 unwritten)
__global__ void k_xT(const float* __restrict__ act, bf16* __restrict__ xT) {
  int idx = blockIdx.x * 256 + threadIdx.x;
  if (idx >= 8 * 512 * 128) return;
  int b = idx >> 16, d = (idx >> 7) & 511, t = idx & 127;
  xT[(t * 16 + b) * 512 + d] = (bf16)act[idx];
}

// teacher-forced prenet input: melTb [t][16][96], k>=80 or t==0 -> 0
__global__ void k_melprep(const float* __restrict__ mel, bf16* __restrict__ melTb) {
  int idx = blockIdx.x * 256 + threadIdx.x;
  if (idx >= 200 * 8 * 96) return;
  int k = idx % 96;
  int b = (idx / 96) % 8;
  int t = idx / (96 * 8);
  bf16 v = (bf16)0.0f;
  if (k < 80 && t > 0) v = (bf16)mel[(b * 200 + (t - 1)) * 80 + k];
  melTb[(t * 16 + b) * 96 + k] = v;
}

// ---------------- encoder ----------------

// conv1d+BN+ReLU as GEMM: M=512, N=1024, K=2560
__global__ void k_conv_enc(const bf16* __restrict__ Wb, const bf16* __restrict__ colb,
                           const float* __restrict__ bias, const float* __restrict__ g,
                           const float* __restrict__ be, float* __restrict__ actout) {
  int m0 = (blockIdx.x & 31) * 16;
  int n0 = (blockIdx.x >> 5) * 16;
  int lane = threadIdx.x & 31, col = lane & 15, half = lane >> 4;
  CFrag fr = gemm_fast(Wb + (m0 + col) * 2560, colb + (n0 + col) * 2560, 2560, lane);
  const float inv = rsqrtf(1.0f + 1e-5f);
#pragma unroll
  for (int r = 0; r < 8; ++r) {
    int m = m0 + r + half * 8;
    int p = n0 + col, b = p >> 7, t = p & 127;
    float y = (fr.f[r] + bias[m]) * inv * g[m] + be[m];
    actout[(b * 512 + m) * 128 + t] = fmaxf(y, 0.0f);
  }
}

// persistent single-WG LSTM, 128 steps; gates M=8(pad16), N=2048, K=512+512.
// k0-outer / nt-inner: one A fragment feeds 8 WMMAs into 8 accumulators.
__global__ void __launch_bounds__(512) k_lstm(
    const bf16* __restrict__ xT, const bf16* __restrict__ Wihb,
    const bf16* __restrict__ Whhb, const float* __restrict__ bih,
    const float* __restrict__ bhh, bf16* __restrict__ hallb,
    float* __restrict__ gates, bf16* __restrict__ hbf, float* __restrict__ cbuf,
    int rev) {
  int tid = threadIdx.x;
  int wave = tid >> 5, lane = tid & 31, col = lane & 15, half = lane >> 4;
  for (int i = tid; i < 4096; i += 512) { hbf[i] = (bf16)0.0f; cbuf[i] = 0.0f; }
  __syncthreads();
  int nbase = wave * 128;
  const bf16* wi = Wihb + (nbase + col) * 512;
  const bf16* wh = Whhb + (nbase + col) * 512;
  for (int s = 0; s < 128; ++s) {
    int t = rev ? 127 - s : s;
    const bf16* xrow = xT + (t * 16 + col) * 512;
    const bf16* hrow = hbf + col * 512;
    CFrag fr[8];
#pragma unroll
    for (int nt = 0; nt < 8; ++nt)
#pragma unroll
      for (int i = 0; i < 8; ++i) fr[nt].f[i] = 0.0f;
    for (int k0 = 0; k0 < 512; k0 += 32) {
      v16bf av = load_frag16(xrow + k0, lane);
#pragma unroll
      for (int nt = 0; nt < 8; ++nt)
        fr[nt].v = wmma_bf16(av, load_frag16(wi + nt * 16 * 512 + k0, lane), fr[nt].v);
    }
    for (int k0 = 0; k0 < 512; k0 += 32) {
      v16bf av = load_frag16(hrow + k0, lane);
#pragma unroll
      for (int nt = 0; nt < 8; ++nt)
        fr[nt].v = wmma_bf16(av, load_frag16(wh + nt * 16 * 512 + k0, lane), fr[nt].v);
    }
#pragma unroll
    for (int nt = 0; nt < 8; ++nt)
#pragma unroll
      for (int r = 0; r < 8; ++r) {
        int m = r + half * 8;
        if (m < 8) gates[m * 2048 + nbase + nt * 16 + col] = fr[nt].f[r];
      }
    __syncthreads();
    for (int i = tid; i < 4096; i += 512) {
      int b = i >> 9, j = i & 511;
      const float* gr = gates + b * 2048;
      float gi = gr[j] + bih[j] + bhh[j];
      float gf = gr[512 + j] + bih[512 + j] + bhh[512 + j];
      float gg = gr[1024 + j] + bih[1024 + j] + bhh[1024 + j];
      float go = gr[1536 + j] + bih[1536 + j] + bhh[1536 + j];
      float c = sigm(gf) * cbuf[i] + sigm(gi) * tanhf(gg);
      float h = sigm(go) * tanhf(c);
      cbuf[i] = c;
      hbf[i] = (bf16)h;
      hallb[(b * 128 + t) * 512 + j] = (bf16)h;  // [p=(b,t)][512]
    }
    __syncthreads();
  }
}

// enc = concat(hf,hb) @ Wl^T + bl : M=1024, N=512, K=512+512
__global__ void k_enclin(const bf16* __restrict__ hfb, const bf16* __restrict__ hbb,
                         const bf16* __restrict__ Wb, const float* __restrict__ bias,
                         float* __restrict__ enc, bf16* __restrict__ encb) {
  int nt = blockIdx.x & 31, mt = blockIdx.x >> 5;
  int m0 = mt * 16, n0 = nt * 16;
  int lane = threadIdx.x & 31, col = lane & 15, half = lane >> 4;
  int p = m0 + col, n = n0 + col;
  CFrag fr;
#pragma unroll
  for (int i = 0; i < 8; ++i) fr.f[i] = 0.0f;
  const bf16* wr = Wb + n * 1024;
  for (int k0 = 0; k0 < 512; k0 += 32)
    fr.v = wmma_bf16(load_frag16(hfb + p * 512 + k0, lane), load_frag16(wr + k0, lane), fr.v);
  for (int k0 = 0; k0 < 512; k0 += 32)
    fr.v = wmma_bf16(load_frag16(hbb + p * 512 + k0, lane),
                     load_frag16(wr + 512 + k0, lane), fr.v);
#pragma unroll
  for (int r = 0; r < 8; ++r) {
    int m = m0 + r + half * 8;
    float v = fr.f[r] + bias[n];
    enc[m * 512 + n] = v;
    encb[m * 512 + n] = (bf16)v;
  }
}

// mem = enc @ Wm^T + bm : M=1024, N=512, K=512
__global__ void k_mem(const bf16* __restrict__ encb, const bf16* __restrict__ Wb,
                      const float* __restrict__ bias, float* __restrict__ mem) {
  int nt = blockIdx.x & 31, mt = blockIdx.x >> 5;
  int m0 = mt * 16, n0 = nt * 16;
  int lane = threadIdx.x & 31, col = lane & 15, half = lane >> 4;
  CFrag fr = gemm_fast(encb + (m0 + col) * 512, Wb + (n0 + col) * 512, 512, lane);
#pragma unroll
  for (int r = 0; r < 8; ++r) {
    int m = m0 + r + half * 8, n = n0 + col;
    mem[m * 512 + n] = fr.f[r] + bias[n];
  }
}

// ---------------- decoder ----------------

struct DecP {
  const bf16 *melTb;
  const float *enc, *mem;
  const bf16 *pre1Wb, *pre2Wb;
  const float *pre1b, *pre2b;
  const bf16 *aWihb, *aWhhb;
  const float *abih, *abhh;
  const bf16 *dWihb, *dWhhb;
  const float *dbih, *dbhh;
  const bf16 *qWb;
  const float *qb;
  const float *lcW, *lcB;      // location conv [32][2][31], [32]
  const bf16 *locWb;
  const float *locb;
  const float *vW, *vb;
  const bf16 *melWb;
  const float *melb;
  const float *stopW, *stopb;
  float *catt, *cdec, *hdec;                        // f32 state
  bf16 *pm1b, *pmb, *hattb, *ctxb, *hdecb, *locfb;  // bf16 state (16 rows)
  float *a, *acum, *e, *q, *gates;
  float *melout, *stopout;
};

__global__ void __launch_bounds__(512) k_decoder(DecP P) {
  int tid = threadIdx.x;
  int wave = tid >> 5, lane = tid & 31, col = lane & 15, half = lane >> 4;
  for (int i = tid; i < 4096; i += 512) {
    P.catt[i] = 0.0f; P.cdec[i] = 0.0f; P.hdec[i] = 0.0f;
    P.hattb[i] = (bf16)0.0f; P.ctxb[i] = (bf16)0.0f; P.hdecb[i] = (bf16)0.0f;
  }
  for (int i = tid; i < 1024; i += 512) { P.a[i] = 0.0f; P.acum[i] = 0.0f; }
  __syncthreads();

  int nbase = wave * 128;
  const bf16* aWi = P.aWihb + (nbase + col) * 768;
  const bf16* aWh = P.aWhhb + (nbase + col) * 512;
  const bf16* dWi = P.dWihb + (nbase + col) * 1024;
  const bf16* dWh = P.dWhhb + (nbase + col) * 512;
  const bf16* qB = P.qWb + (wave * 32 + col) * 512;

  for (int t = 0; t < 200; ++t) {
    // prenet layer 1: K=96 (padded)
    {
      int n = wave * 16 + col;
      CFrag fr = gemm_fast(P.melTb + (t * 16 + col) * 96, P.pre1Wb + n * 96, 96, lane);
#pragma unroll
      for (int r = 0; r < 8; ++r) {
        int m = r + half * 8;
        if (m < 8) P.pm1b[m * 256 + n] = (bf16)fmaxf(fr.f[r] + P.pre1b[n], 0.0f);
      }
    }
    __syncthreads();
    // prenet layer 2: K=256
    {
      int n = wave * 16 + col;
      CFrag fr = gemm_fast(P.pm1b + col * 256, P.pre2Wb + n * 256, 256, lane);
#pragma unroll
      for (int r = 0; r < 8; ++r) {
        int m = r + half * 8;
        if (m < 8) P.pmb[m * 256 + n] = (bf16)fmaxf(fr.f[r] + P.pre2b[n], 0.0f);
      }
    }
    __syncthreads();
    // attention-RNN gates: K = 256(pm)+512(ctx)+512(h_att); 8 acc per wave
    {
      CFrag fr[8];
#pragma unroll
      for (int nt = 0; nt < 8; ++nt)
#pragma unroll
        for (int i = 0; i < 8; ++i) fr[nt].f[i] = 0.0f;
      for (int k0 = 0; k0 < 256; k0 += 32) {
        v16bf av = load_frag16(P.pmb + col * 256 + k0, lane);
#pragma unroll
        for (int nt = 0; nt < 8; ++nt)
          fr[nt].v = wmma_bf16(av, load_frag16(aWi + nt * 16 * 768 + k0, lane), fr[nt].v);
      }
      for (int k0 = 0; k0 < 512; k0 += 32) {
        v16bf av = load_frag16(P.ctxb + col * 512 + k0, lane);
#pragma unroll
        for (int nt = 0; nt < 8; ++nt)
          fr[nt].v =
              wmma_bf16(av, load_frag16(aWi + nt * 16 * 768 + 256 + k0, lane), fr[nt].v);
      }
      for (int k0 = 0; k0 < 512; k0 += 32) {
        v16bf av = load_frag16(P.hattb + col * 512 + k0, lane);
#pragma unroll
        for (int nt = 0; nt < 8; ++nt)
          fr[nt].v = wmma_bf16(av, load_frag16(aWh + nt * 16 * 512 + k0, lane), fr[nt].v);
      }
#pragma unroll
      for (int nt = 0; nt < 8; ++nt)
#pragma unroll
        for (int r = 0; r < 8; ++r) {
          int m = r + half * 8;
          if (m < 8) P.gates[m * 2048 + nbase + nt * 16 + col] = fr[nt].f[r];
        }
    }
    __syncthreads();
    for (int i = tid; i < 4096; i += 512) {
      int b = i >> 9, j = i & 511;
      const float* gr = P.gates + b * 2048;
      float gi = gr[j] + P.abih[j] + P.abhh[j];
      float gf = gr[512 + j] + P.abih[512 + j] + P.abhh[512 + j];
      float gg = gr[1024 + j] + P.abih[1024 + j] + P.abhh[1024 + j];
      float go = gr[1536 + j] + P.abih[1536 + j] + P.abhh[1536 + j];
      float c = sigm(gf) * P.catt[i] + sigm(gi) * tanhf(gg);
      P.catt[i] = c;
      P.hattb[i] = (bf16)(sigm(go) * tanhf(c));
    }
    __syncthreads();
    // q = h_att @ Wq^T + bq (K=512, 2 tiles/wave); location conv -> locf bf16
    {
      CFrag fr[2];
#pragma unroll
      for (int nt = 0; nt < 2; ++nt)
#pragma unroll
        for (int i = 0; i < 8; ++i) fr[nt].f[i] = 0.0f;
      for (int k0 = 0; k0 < 512; k0 += 32) {
        v16bf av = load_frag16(P.hattb + col * 512 + k0, lane);
        fr[0].v = wmma_bf16(av, load_frag16(qB + k0, lane), fr[0].v);
        fr[1].v = wmma_bf16(av, load_frag16(qB + 16 * 512 + k0, lane), fr[1].v);
      }
#pragma unroll
      for (int nt = 0; nt < 2; ++nt)
#pragma unroll
        for (int r = 0; r < 8; ++r) {
          int m = r + half * 8, n = wave * 32 + nt * 16 + col;
          if (m < 8) P.q[m * 512 + n] = fr[nt].f[r] + P.qb[n];
        }
    }
    for (int i = tid; i < 8 * 128 * 32; i += 512) {
      int b = i >> 12, tt = (i >> 5) & 127, c = i & 31;
      float s = P.lcB[c];
      for (int kt = 0; kt < 31; ++kt) {
        int t2 = tt + kt - 15;
        if (t2 >= 0 && t2 < 128)
          s += P.a[b * 128 + t2] * P.lcW[c * 62 + kt] +
               P.acum[b * 128 + t2] * P.lcW[c * 62 + 31 + kt];
      }
      P.locfb[(b * 128 + tt) * 32 + c] = (bf16)s;
    }
    __syncthreads();
    // energies: shared A fragment (K=32) feeds all 32 N-tiles
    for (int mi = 0; mi < 4; ++mi) {
      int p0 = (wave * 4 + mi) * 16;
      v16bf av = load_frag16(P.locfb + (p0 + col) * 32, lane);
      float ep[8];
#pragma unroll
      for (int r = 0; r < 8; ++r) ep[r] = 0.0f;
      for (int nt2 = 0; nt2 < 32; ++nt2) {
        int n0 = nt2 * 16;
        CFrag fr;
#pragma unroll
        for (int i = 0; i < 8; ++i) fr.f[i] = 0.0f;
        fr.v = wmma_bf16(av, load_frag16(P.locWb + (n0 + col) * 32, lane), fr.v);
#pragma unroll
        for (int r = 0; r < 8; ++r) {
          int p = p0 + r + half * 8, b = p >> 7, n = n0 + col;
          float lv = fr.f[r] + P.locb[n];
          ep[r] += tanhf(P.q[b * 512 + n] + P.mem[p * 512 + n] + lv) * P.vW[n];
        }
      }
#pragma unroll
      for (int r = 0; r < 8; ++r) {
        for (int off = 1; off < 16; off <<= 1) ep[r] += __shfl_xor(ep[r], off, 32);
        if (col == 0) P.e[p0 + r + half * 8] = ep[r] + P.vb[0];
      }
    }
    __syncthreads();
    // softmax + a_cum
    if (wave < 8) {
      int b = wave;
      float mx = -1e30f;
      for (int j = lane; j < 128; j += 32) mx = fmaxf(mx, P.e[b * 128 + j]);
      for (int off = 1; off < 32; off <<= 1) mx = fmaxf(mx, __shfl_xor(mx, off, 32));
      float s = 0.0f;
      for (int j = lane; j < 128; j += 32) s += expf(P.e[b * 128 + j] - mx);
      for (int off = 1; off < 32; off <<= 1) s += __shfl_xor(s, off, 32);
      float inv = 1.0f / s;
      for (int j = lane; j < 128; j += 32) {
        float av = expf(P.e[b * 128 + j] - mx) * inv;
        P.a[b * 128 + j] = av;
        P.acum[b * 128 + j] += av;
      }
    }
    __syncthreads();
    // ctx[b][d] = sum_t a[b][t] enc[b][t][d]
    for (int i = tid; i < 4096; i += 512) {
      int b = i >> 9, d = i & 511;
      float s = 0.0f;
      for (int j = 0; j < 128; ++j)
        s += P.a[b * 128 + j] * P.enc[(b * 128 + j) * 512 + d];
      P.ctxb[i] = (bf16)s;
    }
    __syncthreads();
    // decoder-RNN gates: K = 512+512+512; 8 acc per wave
    {
      CFrag fr[8];
#pragma unroll
      for (int nt = 0; nt < 8; ++nt)
#pragma unroll
        for (int i = 0; i < 8; ++i) fr[nt].f[i] = 0.0f;
      for (int k0 = 0; k0 < 512; k0 += 32) {
        v16bf av = load_frag16(P.hattb + col * 512 + k0, lane);
#pragma unroll
        for (int nt = 0; nt < 8; ++nt)
          fr[nt].v = wmma_bf16(av, load_frag16(dWi + nt * 16 * 1024 + k0, lane), fr[nt].v);
      }
      for (int k0 = 0; k0 < 512; k0 += 32) {
        v16bf av = load_frag16(P.ctxb + col * 512 + k0, lane);
#pragma unroll
        for (int nt = 0; nt < 8; ++nt)
          fr[nt].v =
              wmma_bf16(av, load_frag16(dWi + nt * 16 * 1024 + 512 + k0, lane), fr[nt].v);
      }
      for (int k0 = 0; k0 < 512; k0 += 32) {
        v16bf av = load_frag16(P.hdecb + col * 512 + k0, lane);
#pragma unroll
        for (int nt = 0; nt < 8; ++nt)
          fr[nt].v = wmma_bf16(av, load_frag16(dWh + nt * 16 * 512 + k0, lane), fr[nt].v);
      }
#pragma unroll
      for (int nt = 0; nt < 8; ++nt)
#pragma unroll
        for (int r = 0; r < 8; ++r) {
          int m = r + half * 8;
          if (m < 8) P.gates[m * 2048 + nbase + nt * 16 + col] = fr[nt].f[r];
        }
    }
    __syncthreads();
    for (int i = tid; i < 4096; i += 512) {
      int b = i >> 9, j = i & 511;
      const float* gr = P.gates + b * 2048;
      float gi = gr[j] + P.dbih[j] + P.dbhh[j];
      float gf = gr[512 + j] + P.dbih[512 + j] + P.dbhh[512 + j];
      float gg = gr[1024 + j] + P.dbih[1024 + j] + P.dbhh[1024 + j];
      float go = gr[1536 + j] + P.dbih[1536 + j] + P.dbhh[1536 + j];
      float c = sigm(gf) * P.cdec[i] + sigm(gi) * tanhf(gg);
      float h = sigm(go) * tanhf(c);
      P.cdec[i] = c;
      P.hdec[i] = h;
      P.hdecb[i] = (bf16)h;
    }
    __syncthreads();
    // outputs: mel (K=512, N=80 exact = 5 tiles) + stop scalar
    if (wave < 5) {
      int n = wave * 16 + col;
      CFrag fr = gemm_fast(P.hdecb + col * 512, P.melWb + n * 512, 512, lane);
#pragma unroll
      for (int r = 0; r < 8; ++r) {
        int m = r + half * 8;
        if (m < 8) P.melout[(m * 200 + t) * 80 + n] = fr.f[r] + P.melb[n];
      }
    } else if (wave == 5 && lane < 8) {
      float s = P.stopb[0];
      for (int k = 0; k < 512; ++k) s += P.hdec[lane * 512 + k] * P.stopW[k];
      P.stopout[lane * 200 + t] = s;
    }
    __syncthreads();
  }
}

// ---------------- postnet ----------------

// yp (f32) -> im2col bf16 [n=(b,t)][k=cin*5+kt], padded 400->416
__global__ void k_im2col_post(const float* __restrict__ in, int tmajor,
                              bf16* __restrict__ colb) {
  int n = blockIdx.y;
  int b = n / 200, t = n % 200;
  int k = blockIdx.x * 256 + threadIdx.x;
  if (k >= 416) return;
  bf16 v = (bf16)0.0f;
  if (k < 400) {
    int cin = k / 5, kt = k % 5, tt = t + kt - 2;
    if (tt >= 0 && tt < 200)
      v = tmajor ? (bf16)in[(b * 200 + tt) * 80 + cin]
                 : (bf16)in[(b * 80 + cin) * 200 + tt];
  }
  colb[n * 416 + k] = v;
}

// conv1d(80->80,k=5): M=80 (5 tiles), N=1600, K=416 (padded)
__global__ void k_postconv(const bf16* __restrict__ Wb, const bf16* __restrict__ colb,
                           const float* __restrict__ bias, float* __restrict__ out) {
  int m0 = (blockIdx.x % 5) * 16;
  int n0 = (blockIdx.x / 5) * 16;
  int lane = threadIdx.x & 31, col = lane & 15, half = lane >> 4;
  CFrag fr = gemm_fast(Wb + (m0 + col) * 416, colb + (n0 + col) * 416, 416, lane);
#pragma unroll
  for (int r = 0; r < 8; ++r) {
    int m = m0 + r + half * 8, p = n0 + col;
    int b = p / 200, t = p % 200;
    out[(b * 80 + m) * 200 + t] = fr.f[r] + bias[m];
  }
}

__global__ void k_instnorm(float* __restrict__ y, const float* __restrict__ g,
                           const float* __restrict__ be) {
  int b = blockIdx.x / 80, c = blockIdx.x % 80;
  float* row = y + (b * 80 + c) * 200;
  int lane = threadIdx.x & 31;
  float s = 0.0f, s2 = 0.0f;
  for (int j = lane; j < 200; j += 32) { float v = row[j]; s += v; s2 += v * v; }
  for (int off = 1; off < 32; off <<= 1) {
    s += __shfl_xor(s, off, 32);
    s2 += __shfl_xor(s2, off, 32);
  }
  float mu = s * (1.0f / 200.0f);
  float var = s2 * (1.0f / 200.0f) - mu * mu;
  float rs = rsqrtf(var + 1e-5f);
  for (int j = lane; j < 200; j += 32)
    row[j] = tanhf((row[j] - mu) * rs * g[c] + be[c]);
}

// 1x1 conv input pack: colb2 [1600][96]
__global__ void k_im2col_post2(const float* __restrict__ in, bf16* __restrict__ colb2) {
  int n = blockIdx.y;
  int b = n / 200, t = n % 200;
  int k = threadIdx.x;
  if (k >= 96) return;
  colb2[n * 96 + k] = (k < 80) ? (bf16)in[(b * 80 + k) * 200 + t] : (bf16)0.0f;
}

// 1x1 conv (K=96 padded) + residual -> [B][T][M]
__global__ void k_postlast(const bf16* __restrict__ Wb, const bf16* __restrict__ colb2,
                           const float* __restrict__ bias, const float* __restrict__ res,
                           float* __restrict__ out) {
  int m0 = (blockIdx.x % 5) * 16;
  int n0 = (blockIdx.x / 5) * 16;
  int lane = threadIdx.x & 31, col = lane & 15, half = lane >> 4;
  CFrag fr = gemm_fast(Wb + (m0 + col) * 96, colb2 + (n0 + col) * 96, 96, lane);
#pragma unroll
  for (int r = 0; r < 8; ++r) {
    int m = m0 + r + half * 8, p = n0 + col;
    int b = p / 200, t = p % 200;
    int oi = (b * 200 + t) * 80 + m;
    out[oi] = fr.f[r] + bias[m] + res[oi];
  }
}

// ---------------- launch ----------------

extern "C" void kernel_launch(void* const* d_in, const int* in_sizes, int n_in,
                              void* d_out, int out_size, void* d_ws, size_t ws_size,
                              hipStream_t stream) {
  (void)in_sizes; (void)n_in; (void)out_size; (void)ws_size;
  auto F = [&](int i) { return (const float*)d_in[i]; };
  const int* tok = (const int*)d_in[0];
  const float* mel = F(1);
  const float* attconv_w = F(2);   const float* attconv_b = F(3);
  const float* attloc_W  = F(4);   const float* attloc_b  = F(5);
  const float* attm_W    = F(6);   const float* attm_b    = F(7);
  const float* attq_W    = F(8);   const float* attq_b    = F(9);
  const float* arnn_Wih  = F(10);  const float* arnn_Whh  = F(11);
  const float* arnn_bih  = F(12);  const float* arnn_bhh  = F(13);
  const float* attv_W    = F(14);  const float* attv_b    = F(15);
  const float* drnn_Wih  = F(16);  const float* drnn_Whh  = F(17);
  const float* drnn_bih  = F(18);  const float* drnn_bhh  = F(19);
  const float* emb       = F(20);
  const float* ecw[3]  = {F(21), F(25), F(29)};
  const float* ecb[3]  = {F(22), F(26), F(30)};
  const float* ecg[3]  = {F(23), F(27), F(31)};
  const float* ecbe[3] = {F(24), F(28), F(32)};
  const float* encl_W = F(33);  const float* encl_b = F(34);
  const float* melW   = F(35);  const float* melb   = F(36);
  const float* stopW  = F(37);  const float* stopb  = F(38);
  const float* lb_Wih = F(39);  const float* lb_Whh = F(40);
  const float* lb_bih = F(41);  const float* lb_bhh = F(42);
  const float* lf_Wih = F(43);  const float* lf_Whh = F(44);
  const float* lf_bih = F(45);  const float* lf_bhh = F(46);
  const float* pw[4]  = {F(47), F(51), F(55), F(59)};
  const float* pb[4]  = {F(48), F(52), F(56), F(60)};
  const float* pg[4]  = {F(49), F(53), F(57), F(61)};
  const float* pbe[4] = {F(50), F(54), F(58), F(62)};
  const float* plW = F(63);  const float* plb = F(64);
  const float* p1W = F(65);  const float* p1b = F(66);
  const float* p2W = F(67);  const float* p2b = F(68);

  char* base = (char*)d_ws;
  size_t o = 0;
  auto allocf = [&](size_t n) {
    float* p = (float*)(base + o);
    o += (n * 4 + 255) & ~(size_t)255;
    return p;
  };
  auto allocb = [&](size_t n) {
    bf16* p = (bf16*)(base + o);
    o += (n * 2 + 255) & ~(size_t)255;
    return p;
  };
  // f32 buffers
  float* act0 = allocf(524288);
  float* act1 = allocf(524288);
  float* act2 = allocf(524288);
  float* act3 = allocf(524288);
  float* enc = allocf(524288);
  float* mem = allocf(524288);
  float* gatesL = allocf(16384);
  float* cL = allocf(4096);
  float* gates_ = allocf(16384);
  float* catt = allocf(4096);
  float* cdec = allocf(4096);
  float* hdec = allocf(4096);
  float* a_ = allocf(1024);
  float* acum_ = allocf(1024);
  float* e_ = allocf(1024);
  float* q_ = allocf(4096);
  float* yp0 = allocf(128000);
  float* yp1 = allocf(128000);
  // bf16 weights
  bf16* ecwb[3] = {allocb(1310720), allocb(1310720), allocb(1310720)};
  bf16* lfWihb = allocb(1048576);
  bf16* lfWhhb = allocb(1048576);
  bf16* lbWihb = allocb(1048576);
  bf16* lbWhhb = allocb(1048576);
  bf16* enclWb = allocb(524288);
  bf16* attmWb = allocb(262144);
  bf16* attqWb = allocb(262144);
  bf16* aWihb = allocb(1572864);
  bf16* aWhhb = allocb(1048576);
  bf16* dWihb = allocb(2097152);
  bf16* dWhhb = allocb(1048576);
  bf16* pre1Wb = allocb(24576);   // [256][96]
  bf16* pre2Wb = allocb(65536);
  bf16* attlocWb = allocb(16384);
  bf16* melWb = allocb(40960);
  bf16* pwb[4] = {allocb(33280), allocb(33280), allocb(33280), allocb(33280)};
  bf16* plWb = allocb(7680);      // [80][96]
  // bf16 activations
  bf16* colbE = allocb(2621440);  // [1024][2560]
  bf16* xTb = allocb(1048576);    // [128][16][512]
  bf16* hfb = allocb(524288);     // [1024][512]
  bf16* hbb = allocb(524288);
  bf16* encb = allocb(524288);
  bf16* melTb = allocb(307200);   // [200][16][96]
  bf16* hbfL = allocb(8192);      // [16][512]
  bf16* pm1b = allocb(4096);      // [16][256]
  bf16* pmb = allocb(4096);
  bf16* hattb = allocb(8192);
  bf16* ctxb = allocb(8192);
  bf16* hdecb = allocb(8192);
  bf16* locfb = allocb(32768);    // [1024][32]
  bf16* colbP = allocb(665600);   // [1600][416]
  bf16* colbP2 = allocb(153600);  // [1600][96]

  float* out = (float*)d_out;
  float* mel_out = out;
  float* mel_post = out + 128000;
  float* stop_out = out + 256000;

  auto cvt = [&](const float* s, bf16* d, int rows, int cols, int colsPad) {
    dim3 gr((colsPad + 255) / 256, rows);
    k_cvt_pad<<<gr, 256, 0, stream>>>(s, d, cols, colsPad);
  };
  for (int i = 0; i < 3; ++i)
    k_cvt_convw<<<dim3(10, 512), 256, 0, stream>>>(ecw[i], ecwb[i]);
  cvt(lf_Wih, lfWihb, 2048, 512, 512);
  cvt(lf_Whh, lfWhhb, 2048, 512, 512);
  cvt(lb_Wih, lbWihb, 2048, 512, 512);
  cvt(lb_Whh, lbWhhb, 2048, 512, 512);
  cvt(encl_W, enclWb, 512, 1024, 1024);
  cvt(attm_W, attmWb, 512, 512, 512);
  cvt(attq_W, attqWb, 512, 512, 512);
  cvt(arnn_Wih, aWihb, 2048, 768, 768);
  cvt(arnn_Whh, aWhhb, 2048, 512, 512);
  cvt(drnn_Wih, dWihb, 2048, 1024, 1024);
  cvt(drnn_Whh, dWhhb, 2048, 512, 512);
  cvt(p1W, pre1Wb, 256, 80, 96);
  cvt(p2W, pre2Wb, 256, 256, 256);
  cvt(attloc_W, attlocWb, 512, 32, 32);
  cvt(melW, melWb, 80, 512, 512);
  for (int i = 0; i < 4; ++i) cvt(pw[i], pwb[i], 80, 400, 416);
  cvt(plW, plWb, 80, 80, 96);

  k_embed<<<2048, 256, 0, stream>>>(tok, emb, act0);
  const float* acts[4] = {act0, act1, act2, act3};
  for (int i = 0; i < 3; ++i) {
    k_im2col_enc<<<dim3(10, 1024), 256, 0, stream>>>(acts[i], colbE);
    k_conv_enc<<<2048, 32, 0, stream>>>(ecwb[i], colbE, ecb[i], ecg[i], ecbe[i],
                                        (float*)acts[i + 1]);
  }
  k_xT<<<2048, 256, 0, stream>>>(act3, xTb);
  k_lstm<<<1, 512, 0, stream>>>(xTb, lfWihb, lfWhhb, lf_bih, lf_bhh, hfb, gatesL,
                                hbfL, cL, 0);
  k_lstm<<<1, 512, 0, stream>>>(xTb, lbWihb, lbWhhb, lb_bih, lb_bhh, hbb, gatesL,
                                hbfL, cL, 1);
  k_enclin<<<2048, 32, 0, stream>>>(hfb, hbb, enclWb, encl_b, enc, encb);
  k_mem<<<2048, 32, 0, stream>>>(encb, attmWb, attm_b, mem);
  k_melprep<<<600, 256, 0, stream>>>(mel, melTb);

  DecP P;
  P.melTb = melTb; P.enc = enc; P.mem = mem;
  P.pre1Wb = pre1Wb; P.pre2Wb = pre2Wb; P.pre1b = p1b; P.pre2b = p2b;
  P.aWihb = aWihb; P.aWhhb = aWhhb; P.abih = arnn_bih; P.abhh = arnn_bhh;
  P.dWihb = dWihb; P.dWhhb = dWhhb; P.dbih = drnn_bih; P.dbhh = drnn_bhh;
  P.qWb = attqWb; P.qb = attq_b;
  P.lcW = attconv_w; P.lcB = attconv_b;
  P.locWb = attlocWb; P.locb = attloc_b;
  P.vW = attv_W; P.vb = attv_b;
  P.melWb = melWb; P.melb = melb;
  P.stopW = stopW; P.stopb = stopb;
  P.catt = catt; P.cdec = cdec; P.hdec = hdec;
  P.pm1b = pm1b; P.pmb = pmb; P.hattb = hattb; P.ctxb = ctxb; P.hdecb = hdecb;
  P.locfb = locfb;
  P.a = a_; P.acum = acum_; P.e = e_; P.q = q_; P.gates = gates_;
  P.melout = mel_out; P.stopout = stop_out;
  k_decoder<<<1, 512, 0, stream>>>(P);

  k_im2col_post<<<dim3(2, 1600), 256, 0, stream>>>(mel_out, 1, colbP);
  k_postconv<<<500, 32, 0, stream>>>(pwb[0], colbP, pb[0], yp0);
  k_instnorm<<<640, 32, 0, stream>>>(yp0, pg[0], pbe[0]);
  k_im2col_post<<<dim3(2, 1600), 256, 0, stream>>>(yp0, 0, colbP);
  k_postconv<<<500, 32, 0, stream>>>(pwb[1], colbP, pb[1], yp1);
  k_instnorm<<<640, 32, 0, stream>>>(yp1, pg[1], pbe[1]);
  k_im2col_post<<<dim3(2, 1600), 256, 0, stream>>>(yp1, 0, colbP);
  k_postconv<<<500, 32, 0, stream>>>(pwb[2], colbP, pb[2], yp0);
  k_instnorm<<<640, 32, 0, stream>>>(yp0, pg[2], pbe[2]);
  k_im2col_post<<<dim3(2, 1600), 256, 0, stream>>>(yp0, 0, colbP);
  k_postconv<<<500, 32, 0, stream>>>(pwb[3], colbP, pb[3], yp1);
  k_instnorm<<<640, 32, 0, stream>>>(yp1, pg[3], pbe[3]);
  k_im2col_post2<<<dim3(1, 1600), 128, 0, stream>>>(yp1, colbP2);
  k_postlast<<<500, 32, 0, stream>>>(plWb, colbP2, plb, mel_out, mel_post);
}
